// PopulationGCN_32641751450119
// MI455X (gfx1250) — matrix operand order repaired and checked by reading, compiled, not run on hardware
//
#include <hip/hip_runtime.h>
#include <hip/hip_bf16.h>
#include <math.h>

typedef __attribute__((ext_vector_type(16))) _Float16 v16h;
typedef __attribute__((ext_vector_type(8)))  _Float16 v8h;
typedef __attribute__((ext_vector_type(8)))  float    v8f;

#define N_NODES 20000
#define SEQ_T   24
#define EMB     64
#define HL      128          // H_LSTM
#define HG      128          // H_GCN
#define OUTF    16
#define GATES   (4*HL)       // 512

// ---------------- WMMA fragment helpers (CDNA5 16x16x32 f16 layouts) ----------
// A (16x32, row-major, lda halfs): lane = row (0..15), half = lane>>4
//   e=0..7  -> K = kb + half*8 + e        (contiguous 16B)
//   e=8..15 -> K = kb + 16 + half*8 + e-8 (contiguous 16B)
__device__ __forceinline__ v16h frag_a(const _Float16* A, int lda, int row, int half, int kb) {
    union { v16h v; v8h h[2]; } u;
    const _Float16* p = A + row * lda + kb + half * 8;
    u.h[0] = *(const v8h*)(p);
    u.h[1] = *(const v8h*)(p + 16);
    return u.v;
}
// B (32x16) supplied as Bt = B^T: [N][K] row-major, ldb halfs.
//   lane = col n (0..15), half; e=0..7 -> K = kb + half*8 + e, e=8..15 -> K = kb + 16 + half*8 + e-8
__device__ __forceinline__ v16h frag_b(const _Float16* Bt, int ldb, int col, int half, int kb) {
    union { v16h v; v8h h[2]; } u;
    const _Float16* p = Bt + col * ldb + kb + half * 8;
    u.h[0] = *(const v8h*)(p);
    u.h[1] = *(const v8h*)(p + 16);
    return u.v;
}
__device__ __forceinline__ v8f wmma16(v16h a, v16h b, v8f c) {
    return __builtin_amdgcn_wmma_f32_16x16x32_f16(false, a, false, b, (short)0, c, false, false);
}
__device__ __forceinline__ float sigm(float x) { return 1.0f / (1.0f + __expf(-x)); }

// ---------------- prep: fp16 weight copies (pre-transposed to Bt=[N][K]) ------
__global__ __launch_bounds__(256) void k_prep(
    const float* __restrict__ W_ih, const float* __restrict__ W_hh,
    const float* __restrict__ b_ih, const float* __restrict__ b_hh,
    const float* __restrict__ c1W,  const float* __restrict__ c2W,
    const float* __restrict__ fcW,
    _Float16* wih16, _Float16* whh16, float* bias,
    _Float16* w1t, _Float16* w2t, _Float16* fct)
{
    int i = blockIdx.x * 256 + threadIdx.x;
    const int S0 = GATES * EMB;        // Bt for x@W_ih^T is W_ih itself: [512][64]
    const int S1 = S0 + GATES * HL;    // W_hh: [512][128]
    const int S2 = S1 + GATES;         // combined bias
    const int S3 = S2 + HG * HL;       // conv1_W^T
    const int S4 = S3 + HG * HG;       // conv2_W^T
    const int S5 = S4 + OUTF * HG;     // fc_W^T
    if (i < S0) { wih16[i] = (_Float16)W_ih[i]; return; }
    if (i < S1) { int j = i - S0; whh16[j] = (_Float16)W_hh[j]; return; }
    if (i < S2) { int j = i - S1; bias[j] = b_ih[j] + b_hh[j]; return; }
    if (i < S3) { int j = i - S2; int n = j / HL, k = j % HL; w1t[n*HL+k] = (_Float16)c1W[k*HG+n]; return; }
    if (i < S4) { int j = i - S3; int n = j / HG, k = j % HG; w2t[n*HG+k] = (_Float16)c2W[k*HG+n]; return; }
    if (i < S5) { int j = i - S4; int n = j / HG, k = j % HG; fct[n*HG+k] = (_Float16)fcW[k*OUTF+n]; return; }
}

__global__ __launch_bounds__(256) void k_emb16(const float* __restrict__ e, _Float16* o, int n) {
    int i = blockIdx.x * 256 + threadIdx.x;
    if (i < n) o[i] = (_Float16)e[i];
}
__global__ __launch_bounds__(256) void k_deg_init(float* deg, int n) {
    int i = blockIdx.x * 256 + threadIdx.x;
    if (i < n) deg[i] = 1.0f;                       // self-loop weight
}
__global__ __launch_bounds__(256) void k_deg_acc(const int* __restrict__ dst,
                                                 const float* __restrict__ w, float* deg, int n) {
    int i = blockIdx.x * 256 + threadIdx.x;
    if (i < n) atomicAdd(&deg[dst[i]], w[i]);
}
__global__ __launch_bounds__(256) void k_rsqrt(float* d, int n) {
    int i = blockIdx.x * 256 + threadIdx.x;
    if (i < n) d[i] = rsqrtf(d[i]);
}
__global__ __launch_bounds__(256) void k_zero(float* p, int n) {
    int i = blockIdx.x * 256 + threadIdx.x;
    if (i < n) p[i] = 0.0f;
}

// ---------------- fused embedding-gather + 24-step LSTM ------------------------
// Block: 256 threads (8 wave32), 32 nodes. LDS: Wih(64K)+Whh(128K)+H(8K)+X(4K)=204K.
// Each wave owns a fixed (node-half, 32-wide hidden slice) of the state for all
// timesteps: the 4 gate tiles land in its own WMMA accumulators (C-layout), so the
// cell state c stays in registers and only h is written back to LDS (fp16 A-layout).
__global__ __launch_bounds__(256) void k_lstm(
    const int* __restrict__ tokens, const int* __restrict__ lengths,
    const _Float16* __restrict__ emb16,
    const _Float16* __restrict__ wih, const _Float16* __restrict__ whh,
    const float* __restrict__ bias, _Float16* __restrict__ hlast)
{
    extern __shared__ char smem_raw[];
    _Float16* sWih = (_Float16*)smem_raw;          // [512][64]  Bt layout
    _Float16* sWhh = sWih + GATES * EMB;           // [512][128] Bt layout
    _Float16* sH   = sWhh + GATES * HL;            // [32][128]  A (row-major)
    _Float16* sX   = sH + 32 * HL;                 // [32][64]   A (row-major)

    const int tid = threadIdx.x;
    const int node0 = blockIdx.x * 32;
    const int wave = tid >> 5, lane = tid & 31;
    const int row = lane & 15, half = lane >> 4;
    const int mt  = wave >> 2;                     // node half-block: 0..1
    const int jt0 = (wave & 3) * 2;                // hidden j-tiles {jt0, jt0+1}

    // weights -> LDS (b128 copies), zero h0
    for (int i = tid; i < GATES * EMB / 8; i += 256) ((uint4*)sWih)[i] = ((const uint4*)wih)[i];
    for (int i = tid; i < GATES * HL  / 8; i += 256) ((uint4*)sWhh)[i] = ((const uint4*)whh)[i];
    for (int i = tid; i < 32 * HL; i += 256) sH[i] = (_Float16)0.0f;
    // gather x_0 : 32 nodes x 64 halfs = 256 x 16B chunks
    {
        int ni = tid >> 3, c = tid & 7;
        int tok = tokens[(node0 + ni) * SEQ_T];
        ((uint4*)sX)[ni * 8 + c] = ((const uint4*)(emb16 + tok * EMB))[c];
    }
    // lengths for this wave's node rows (C-layout: node = mt*16 + half*8 + r)
    int lenr[8];
    #pragma unroll
    for (int r = 0; r < 8; ++r) lenr[r] = lengths[node0 + mt * 16 + half * 8 + r];

    float cst[2][8];                               // persistent cell state (C-layout)
    #pragma unroll
    for (int p = 0; p < 2; ++p)
        #pragma unroll
        for (int r = 0; r < 8; ++r) cst[p][r] = 0.0f;

    __syncthreads();

    for (int t = 0; t < SEQ_T; ++t) {
        // A fragments shared by all 4 gates and both j-tiles of this wave
        const _Float16* Ax = sX + mt * 16 * EMB;
        const _Float16* Ah = sH + mt * 16 * HL;
        v16h ax0 = frag_a(Ax, EMB, row, half, 0);
        v16h ax1 = frag_a(Ax, EMB, row, half, 32);
        v16h ah0 = frag_a(Ah, HL, row, half, 0);
        v16h ah1 = frag_a(Ah, HL, row, half, 32);
        v16h ah2 = frag_a(Ah, HL, row, half, 64);
        v16h ah3 = frag_a(Ah, HL, row, half, 96);

        float hnew[2][8];
        #pragma unroll
        for (int p = 0; p < 2; ++p) {
            int jt = jt0 + p;
            float zg4[4][8];
            #pragma unroll
            for (int g = 0; g < 4; ++g) {          // PyTorch gate order i,f,g,o
                const _Float16* Bi = sWih + (g * HL + jt * 16) * EMB;
                const _Float16* Bh = sWhh + (g * HL + jt * 16) * HL;
                v8f acc = {};
                acc = wmma16(ax0, frag_b(Bi, EMB, row, half, 0),  acc);
                acc = wmma16(ax1, frag_b(Bi, EMB, row, half, 32), acc);
                acc = wmma16(ah0, frag_b(Bh, HL, row, half, 0),   acc);
                acc = wmma16(ah1, frag_b(Bh, HL, row, half, 32),  acc);
                acc = wmma16(ah2, frag_b(Bh, HL, row, half, 64),  acc);
                acc = wmma16(ah3, frag_b(Bh, HL, row, half, 96),  acc);
                float bg = bias[g * HL + jt * 16 + row];
                #pragma unroll
                for (int r = 0; r < 8; ++r) zg4[g][r] = acc[r] + bg;
            }
            #pragma unroll
            for (int r = 0; r < 8; ++r) {
                float cc = sigm(zg4[1][r]) * cst[p][r] + sigm(zg4[0][r]) * tanhf(zg4[2][r]);
                cst[p][r] = cc;
                hnew[p][r] = sigm(zg4[3][r]) * tanhf(cc);
            }
        }
        __syncthreads();                           // all sX/sH reads complete
        // write-phase: h -> sH (A layout), gather x_{t+1}, emit selected rows
        #pragma unroll
        for (int p = 0; p < 2; ++p) {
            int jt = jt0 + p;
            #pragma unroll
            for (int r = 0; r < 8; ++r) {
                int nl = mt * 16 + half * 8 + r;
                _Float16 hv = (_Float16)hnew[p][r];
                sH[nl * HL + jt * 16 + row] = hv;
                if (t == lenr[r] - 1)
                    hlast[(node0 + nl) * HL + jt * 16 + row] = hv;
            }
        }
        if (t + 1 < SEQ_T) {
            int ni = tid >> 3, c = tid & 7;
            int tok = tokens[(node0 + ni) * SEQ_T + t + 1];
            ((uint4*)sX)[ni * 8 + c] = ((const uint4*)(emb16 + tok * EMB))[c];
        }
        __syncthreads();                           // writes visible for next step
    }
}

// ---------------- generic WMMA GEMM: one wave per 16x16 tile ------------------
// out[m][n] (f32, ld = n_tiles*16) = A[m][:K](f16) . Bt[n][:K](f16)  (+ bias[n])
__global__ __launch_bounds__(256) void k_gemm(
    const _Float16* __restrict__ A, const _Float16* __restrict__ Bt,
    float* __restrict__ out, int m_tiles, int n_tiles, int K,
    const float* __restrict__ bias)
{
    int gw = blockIdx.x * 8 + (threadIdx.x >> 5);
    int lane = threadIdx.x & 31, row = lane & 15, half = lane >> 4;
    int mt = gw / n_tiles, nt = gw % n_tiles;
    if (mt >= m_tiles) return;
    const _Float16* Ab = A + mt * 16 * K;
    const _Float16* Bb = Bt + nt * 16 * K;
    v8f acc = {};
    for (int kb = 0; kb < K; kb += 32)
        acc = wmma16(frag_a(Ab, K, row, half, kb), frag_b(Bb, K, row, half, kb), acc);
    int ldo = n_tiles * 16;
    float bv = bias ? bias[nt * 16 + row] : 0.0f;
    float* ob = out + (mt * 16 + half * 8) * ldo + nt * 16 + row;
    #pragma unroll
    for (int r = 0; r < 8; ++r) ob[r * ldo] = acc[r] + bv;
}

// ---------------- GCN edge scatter: one wave per edge, lane = 4 floats --------
__global__ __launch_bounds__(256) void k_scatter(
    const int* __restrict__ src, const int* __restrict__ dst,
    const float* __restrict__ w, const float* __restrict__ dinv,
    const float* __restrict__ feat, float* __restrict__ agg, int n_edge)
{
    int g = blockIdx.x * 256 + threadIdx.x;
    int lane = g & 31, e = g >> 5;
    if (e >= n_edge) return;
    int s = src[e], d = dst[e];
    float norm = dinv[s] * w[e] * dinv[d];
    float4 v = ((const float4*)(feat + s * HG))[lane];
    float* ad = agg + d * HG + lane * 4;
    atomicAdd(ad + 0, v.x * norm);
    atomicAdd(ad + 1, v.y * norm);
    atomicAdd(ad + 2, v.z * norm);
    atomicAdd(ad + 3, v.w * norm);
}

// agg + self-loop + bias -> relu -> fp16 for the next WMMA GEMM
__global__ __launch_bounds__(256) void k_finalize(
    const float* __restrict__ agg, const float* __restrict__ feat,
    const float* __restrict__ dinv, const float* __restrict__ bias,
    _Float16* __restrict__ out16, int n)
{
    int idx = blockIdx.x * 256 + threadIdx.x;
    if (idx >= n) return;
    int i = idx >> 7, j = idx & (HG - 1);
    float di = dinv[i];
    float v = agg[idx] + feat[idx] * di * di + bias[j];
    out16[idx] = (_Float16)fmaxf(v, 0.0f);
}

// ---------------- host orchestration ------------------------------------------
extern "C" void kernel_launch(void* const* d_in, const int* in_sizes, int n_in,
                              void* d_out, int out_size, void* d_ws, size_t ws_size,
                              hipStream_t stream) {
    const int*   x_tokens = (const int*)  d_in[0];
    const int*   lengths  = (const int*)  d_in[1];
    const int*   edge_idx = (const int*)  d_in[2];
    const float* edge_w   = (const float*)d_in[3];
    const float* emb      = (const float*)d_in[4];
    const float* W_ih     = (const float*)d_in[5];
    const float* W_hh     = (const float*)d_in[6];
    const float* b_ih     = (const float*)d_in[7];
    const float* b_hh     = (const float*)d_in[8];
    const float* c1W      = (const float*)d_in[9];
    const float* c1b      = (const float*)d_in[10];
    const float* c2W      = (const float*)d_in[11];
    const float* c2b      = (const float*)d_in[12];
    const float* fcW      = (const float*)d_in[13];
    const float* fcb      = (const float*)d_in[14];
    const int n_edge = in_sizes[3];
    const int emb_elems = in_sizes[4];
    const int* e_src = edge_idx;
    const int* e_dst = edge_idx + n_edge;

    char* ws = (char*)d_ws;
    size_t off = 0;
    auto take = [&](size_t bytes) -> char* {
        char* p = ws + off; off += (bytes + 255) & ~(size_t)255; return p;
    };
    _Float16* emb16  = (_Float16*)take((size_t)emb_elems * 2);
    _Float16* wih16  = (_Float16*)take((size_t)GATES * EMB * 2);
    _Float16* whh16  = (_Float16*)take((size_t)GATES * HL * 2);
    float*    bias   = (float*)   take((size_t)GATES * 4);
    _Float16* w1t    = (_Float16*)take((size_t)HG * HL * 2);
    _Float16* w2t    = (_Float16*)take((size_t)HG * HG * 2);
    _Float16* fct    = (_Float16*)take((size_t)OUTF * HG * 2);
    _Float16* hlast  = (_Float16*)take((size_t)N_NODES * HL * 2);  // reused as x2
    _Float16* x1     = (_Float16*)take((size_t)N_NODES * HG * 2);
    float*    feat   = (float*)   take((size_t)N_NODES * HG * 4);
    float*    agg    = (float*)   take((size_t)N_NODES * HG * 4);
    float*    dinv   = (float*)   take((size_t)N_NODES * 4);

    const int NF = N_NODES * HG;                     // 2,560,000
    const int MT = N_NODES / 16;                     // 1250 m-tiles

    // prep: fp16 weights (+transposes), embedding table, symmetric norm
    k_prep<<<(GATES*EMB + GATES*HL + GATES + HG*HL + HG*HG + OUTF*HG + 255)/256, 256, 0, stream>>>(
        W_ih, W_hh, b_ih, b_hh, c1W, c2W, fcW, wih16, whh16, bias, w1t, w2t, fct);
    k_emb16<<<(emb_elems + 255)/256, 256, 0, stream>>>(emb, emb16, emb_elems);
    k_deg_init<<<(N_NODES + 255)/256, 256, 0, stream>>>(dinv, N_NODES);
    k_deg_acc<<<(n_edge + 255)/256, 256, 0, stream>>>(e_dst, edge_w, dinv, n_edge);
    k_rsqrt<<<(N_NODES + 255)/256, 256, 0, stream>>>(dinv, N_NODES);

    // fused embedding + LSTM (weights-stationary in LDS, state in registers)
    size_t lds_bytes = (size_t)(GATES*EMB + GATES*HL + 32*HL + 32*EMB) * 2;  // 208,896 B
    k_lstm<<<N_NODES/32, 256, lds_bytes, stream>>>(
        x_tokens, lengths, emb16, wih16, whh16, bias, hlast);

    // GCN conv1
    k_gemm<<<(MT*8 + 7)/8, 256, 0, stream>>>(hlast, w1t, feat, MT, HG/16, HL, nullptr);
    k_zero<<<(NF + 255)/256, 256, 0, stream>>>(agg, NF);
    k_scatter<<<(n_edge*32 + 255)/256, 256, 0, stream>>>(e_src, e_dst, edge_w, dinv, feat, agg, n_edge);
    k_finalize<<<(NF + 255)/256, 256, 0, stream>>>(agg, feat, dinv, c1b, x1, NF);

    // GCN conv2
    k_gemm<<<(MT*8 + 7)/8, 256, 0, stream>>>(x1, w2t, feat, MT, HG/16, HG, nullptr);
    k_zero<<<(NF + 255)/256, 256, 0, stream>>>(agg, NF);
    k_scatter<<<(n_edge*32 + 255)/256, 256, 0, stream>>>(e_src, e_dst, edge_w, dinv, feat, agg, n_edge);
    k_finalize<<<(NF + 255)/256, 256, 0, stream>>>(agg, feat, dinv, c2b, hlast /*as x2*/, NF);

    // final FC with bias, f32 output
    k_gemm<<<(MT + 7)/8, 256, 0, stream>>>(hlast, fct, (float*)d_out, MT, OUTF/16, HG, fcb);
}
